// CBR_RNN_87883620811429
// MI455X (gfx1250) — matrix operand order repaired and checked by reading, not compiled, not run on hardware
//
#include <hip/hip_runtime.h>

#define NTOKEN   32000
#define NINP     1024
#define NHID     1024
#define NHEADS   4
#define HEAD_DIM 256
#define SEQ      256
#define BATCH    32

typedef __attribute__((ext_vector_type(16))) __bf16 bf16x16;
typedef __attribute__((ext_vector_type(8)))  __bf16 bf16x8;
typedef __attribute__((ext_vector_type(4)))  __bf16 bf16x4;
typedef __attribute__((ext_vector_type(8)))  float  f32x8;

__device__ __forceinline__ __bf16 to_bf16(float f) { return (__bf16)f; }

// ---------------------------------------------------------------------------
// fp32 -> bf16 weight conversion (one pass; weights then live in L2 as bf16)
// ---------------------------------------------------------------------------
__global__ void k_cvt_bf16(const float* __restrict__ src, __bf16* __restrict__ dst, long n) {
    long i = (long)blockIdx.x * blockDim.x + threadIdx.x;
    long stride = (long)gridDim.x * blockDim.x;
    for (; i < n; i += stride) dst[i] = to_bf16(src[i]);
}

// ---------------------------------------------------------------------------
// embedding gather: emb_bf[row, :] = bf16(emb_W[obs[row], :]), row = s*B + b
// ---------------------------------------------------------------------------
__global__ void k_gather_emb(const float* __restrict__ embW, const int* __restrict__ obs,
                             __bf16* __restrict__ out) {
    int row = blockIdx.x;
    int tok = obs[row];
    const float* s = embW + (long)tok * NINP;
    __bf16* d = out + (long)row * NINP;
    for (int c = threadIdx.x; c < NINP; c += blockDim.x) d[c] = to_bf16(s[c]);
}

// ---------------------------------------------------------------------------
// cache init: keys[0]=key0, values[0]=value0, hidden=hidden0
// ---------------------------------------------------------------------------
__global__ void k_init(const float* __restrict__ k0, const float* __restrict__ v0,
                       const float* __restrict__ h0, float* __restrict__ keys,
                       float* __restrict__ values, float* __restrict__ hidden) {
    int i = blockIdx.x * blockDim.x + threadIdx.x;
    if (i < BATCH * NHID) { keys[i] = k0[i]; values[i] = v0[i]; hidden[i] = h0[i]; }
}

// ---------------------------------------------------------------------------
// per-step input packing: xq = [emb_i | hidden], xi = [emb_i | q | attn | hidden]
// ---------------------------------------------------------------------------
__global__ void k_build_inputs(const __bf16* __restrict__ embRow, const float* __restrict__ hidden,
                               __bf16* __restrict__ xq, __bf16* __restrict__ xi) {
    int i = blockIdx.x * blockDim.x + threadIdx.x;
    if (i >= BATCH * NHID) return;
    int b = i >> 10, c = i & 1023;
    __bf16 e = embRow[i];
    __bf16 h = to_bf16(hidden[i]);
    xq[b * 2048 + c]        = e;
    xq[b * 2048 + 1024 + c] = h;
    xi[b * 4096 + c]        = e;
    xi[b * 4096 + 3072 + c] = h;
}

// ---------------------------------------------------------------------------
// WMMA GEMM:  C[M,N] = A[M,K] (bf16) @ W[N,K]^T (bf16) + bias  (fp32 out)
//   grid.x = N/256, grid.y = M/32, block = 256 threads = 8 waves.
//   Block tile: 32(M) x 256(N); each wave owns 16(M) x 64(N) = 4 accumulators.
//   K in 256-wide chunks, double-buffered in LDS (2 x 16KB): chunk c+1 is
//   fetched with GLOBAL_LOAD_ASYNC_TO_LDS_B128 (ASYNCcnt) while chunk c feeds
//   32 v_wmma_f32_16x16x32_bf16 per wave; the asynccnt wait lands *after* the
//   compute block, hiding the global->LDS latency. B fragments stream straight
//   from the row-major [N,K] weight (contiguous 32B per lane, same k order as the A fragment).
// ---------------------------------------------------------------------------
#define KC 256

__global__ __launch_bounds__(256)
void k_gemm_wmma(const __bf16* __restrict__ A, const __bf16* __restrict__ W,
                 const float* __restrict__ bias, float* __restrict__ C,
                 int K, int N) {
    __shared__ __bf16 lds_a[2 * 32 * KC];   // double buffer, 16KB each

    const int tid   = threadIdx.x;
    const int wave  = tid >> 5;
    const int lane  = tid & 31;
    const int mbase = blockIdx.y * 32;
    const int mt    = (wave & 1) * 16;                       // wave's M sub-tile
    const int ntb   = blockIdx.x * 256 + (wave >> 1) * 64;   // wave's N group
    const int hh    = lane >> 4;                             // half-wave select
    const int l15   = lane & 15;

    // per-lane weight-row base pointers for the 4 N-tiles
    const __bf16* wrow0 = W + (long)(ntb +  0 + l15) * K + hh * 16;
    const __bf16* wrow1 = W + (long)(ntb + 16 + l15) * K + hh * 16;
    const __bf16* wrow2 = W + (long)(ntb + 32 + l15) * K + hh * 16;
    const __bf16* wrow3 = W + (long)(ntb + 48 + l15) * K + hh * 16;

    // this thread's slice of each A-chunk copy: 32 contiguous bf16 (64B)
    const int cr = (tid * 32) >> 8;       // row 0..31
    const int cc = (tid * 32) & 255;      // col, step 32
    const unsigned ldsOff0 = (unsigned)(size_t)&lds_a[cr * KC + cc];
    const unsigned ldsOff1 = ldsOff0 + 32 * KC * 2;
    const __bf16* gsrc = A + (long)(mbase + cr) * K + cc;

    // async global -> LDS copy of one 32xKC chunk slice (4 x b128 per thread)
    #define ASYNC_CHUNK(dstOff, gp)                                          \
        do {                                                                 \
            unsigned long long ga_ = (unsigned long long)(gp);               \
            asm volatile(                                                    \
                "global_load_async_to_lds_b128 %0, %1, off\n\t"              \
                "global_load_async_to_lds_b128 %0, %1, off offset:16\n\t"    \
                "global_load_async_to_lds_b128 %0, %1, off offset:32\n\t"    \
                "global_load_async_to_lds_b128 %0, %1, off offset:48"        \
                :: "v"(dstOff), "v"(ga_) : "memory");                        \
        } while (0)

    f32x8 acc0 = {}, acc1 = {}, acc2 = {}, acc3 = {};

    // prologue: chunk 0 into buffer 0
    ASYNC_CHUNK(ldsOff0, gsrc);
    asm volatile("s_wait_asynccnt 0" ::: "memory");
    __syncthreads();

    const int nchunks = K / KC;
    for (int c = 0; c < nchunks; ++c) {
        const int cur = c & 1;
        // kick off next chunk into the other buffer (overlaps with compute)
        if (c + 1 < nchunks)
            ASYNC_CHUNK(cur ? ldsOff0 : ldsOff1, gsrc + (long)(c + 1) * KC);

        const __bf16* abuf = lds_a + cur * 32 * KC;
        const int kk = c * KC;

        #pragma unroll
        for (int ks = 0; ks < KC / 32; ++ks) {
            // A fragment (16x32): lanes 0-15 K0..15, 16-31 K16..31
            const __bf16* ar = abuf + (mt + l15) * KC + ks * 32 + hh * 16;
            bf16x8 alo = *(const bf16x8*)(ar);
            bf16x8 ahi = *(const bf16x8*)(ar + 8);
            bf16x16 afrag;
            #pragma unroll
            for (int j = 0; j < 8; ++j) { afrag[j] = alo[j]; afrag[j + 8] = ahi[j]; }

            const int ko = kk + ks * 32;
            bf16x16 b0 = *(const bf16x16*)(wrow0 + ko);
            bf16x16 b1 = *(const bf16x16*)(wrow1 + ko);
            bf16x16 b2 = *(const bf16x16*)(wrow2 + ko);
            bf16x16 b3 = *(const bf16x16*)(wrow3 + ko);
            __builtin_prefetch(wrow0 + ko + KC, 0, 3);   // next chunk, keep near

            acc0 = __builtin_amdgcn_wmma_f32_16x16x32_bf16(
                false, afrag, false, b0, (short)0, acc0, false, false);
            acc1 = __builtin_amdgcn_wmma_f32_16x16x32_bf16(
                false, afrag, false, b1, (short)0, acc1, false, false);
            acc2 = __builtin_amdgcn_wmma_f32_16x16x32_bf16(
                false, afrag, false, b2, (short)0, acc2, false, false);
            acc3 = __builtin_amdgcn_wmma_f32_16x16x32_bf16(
                false, afrag, false, b3, (short)0, acc3, false, false);
        }

        // next chunk's LDS writes complete + all waves done reading `cur`
        asm volatile("s_wait_asynccnt 0" ::: "memory");
        __syncthreads();
    }
    #undef ASYNC_CHUNK

    // C/D layout: VGPR r -> row mbase+mt+hh*8+r, column ntile + (lane&15)
    const int row0 = mbase + mt + hh * 8;
    #pragma unroll
    for (int j = 0; j < 4; ++j) {
        int col = ntb + j * 16 + l15;
        float bv = bias ? bias[col] : 0.0f;
        f32x8 a = (j == 0) ? acc0 : (j == 1) ? acc1 : (j == 2) ? acc2 : acc3;
        #pragma unroll
        for (int r = 0; r < 8; ++r)
            C[(long)(row0 + r) * N + col] = a[r] + bv;
    }
}

// ---------------------------------------------------------------------------
// LayerNorm + ReLU over rows; optional fp32 out and/or bf16 out (packed into
// a wider activation buffer at column offset boff with row stride bstride).
// ---------------------------------------------------------------------------
__global__ __launch_bounds__(256)
void k_ln_relu(const float* __restrict__ x, const float* __restrict__ g,
               const float* __restrict__ beta, float* __restrict__ outf,
               __bf16* __restrict__ outb, int N, int bstride, int boff) {
    __shared__ float s1[256], s2[256];
    int row = blockIdx.x;
    const float* xr = x + (long)row * N;
    float a = 0.f, q = 0.f;
    for (int c = threadIdx.x; c < N; c += 256) { float v = xr[c]; a += v; q += v * v; }
    s1[threadIdx.x] = a; s2[threadIdx.x] = q;
    __syncthreads();
    for (int off = 128; off > 0; off >>= 1) {
        if (threadIdx.x < off) {
            s1[threadIdx.x] += s1[threadIdx.x + off];
            s2[threadIdx.x] += s2[threadIdx.x + off];
        }
        __syncthreads();
    }
    float mean = s1[0] / N;
    float var  = s2[0] / N - mean * mean;
    float rstd = rsqrtf(var + 1e-5f);
    for (int c = threadIdx.x; c < N; c += 256) {
        float y = (xr[c] - mean) * rstd * g[c] + beta[c];
        y = fmaxf(y, 0.f);
        if (outf) outf[(long)row * N + c] = y;
        if (outb) outb[(long)row * bstride + boff + c] = to_bf16(y);
    }
}

// ---------------------------------------------------------------------------
// growing-cache attention for step i: one block per (batch, head).
// ---------------------------------------------------------------------------
__global__ __launch_bounds__(256)
void k_attn(const float* __restrict__ qf, const float* __restrict__ keys,
            const float* __restrict__ values, __bf16* __restrict__ xi, int step) {
    __shared__ float qs[HEAD_DIM];
    __shared__ float sc[SEQ + 8];
    __shared__ float red[256];
    int b = blockIdx.x >> 2;
    int h = blockIdx.x & 3;
    int t = threadIdx.x;
    int num = step + 1;

    qs[t] = qf[b * NHID + h * HEAD_DIM + t];
    __syncthreads();

    const float scale = 0.625f;   // 1 / (sqrt(256) * 0.1)
    for (int s = t; s < num; s += 256) {
        const float* kr = keys + ((long)s * BATCH + b) * NHID + h * HEAD_DIM;
        float d = 0.f;
        #pragma unroll 4
        for (int c = 0; c < HEAD_DIM; ++c) d += qs[c] * kr[c];
        sc[s] = d * scale;
    }
    __syncthreads();

    float m = -3.0e38f;
    for (int s = t; s < num; s += 256) m = fmaxf(m, sc[s]);
    red[t] = m; __syncthreads();
    for (int off = 128; off > 0; off >>= 1) {
        if (t < off) red[t] = fmaxf(red[t], red[t + off]);
        __syncthreads();
    }
    m = red[0]; __syncthreads();

    float sum = 0.f;
    for (int s = t; s < num; s += 256) { float e = __expf(sc[s] - m); sc[s] = e; sum += e; }
    red[t] = sum; __syncthreads();
    for (int off = 128; off > 0; off >>= 1) {
        if (t < off) red[t] += red[t + off];
        __syncthreads();
    }
    float inv = 1.0f / red[0];
    __syncthreads();

    float acc = 0.f;
    for (int s = 0; s < num; ++s)
        acc += sc[s] * values[((long)s * BATCH + b) * NHID + h * HEAD_DIM + t];
    xi[b * 4096 + 2048 + h * HEAD_DIM + t] = to_bf16(acc * inv);
}

// ---------------------------------------------------------------------------
// split fin -> (key, value, hidden); append KV cache, record state in bf16
// ---------------------------------------------------------------------------
__global__ void k_commit(const float* __restrict__ fin, float* __restrict__ keys,
                         float* __restrict__ values, float* __restrict__ hidden,
                         __bf16* __restrict__ states, int step) {
    int i = blockIdx.x * blockDim.x + threadIdx.x;
    if (i >= BATCH * NHID) return;
    int b = i >> 10, c = i & 1023;
    const float* fr = fin + (long)b * 3 * NHID;
    keys[((long)(step + 1) * BATCH + b) * NHID + c]   = fr[c];
    values[((long)(step + 1) * BATCH + b) * NHID + c] = fr[NHID + c];
    float hv = fr[2 * NHID + c];
    hidden[i] = hv;
    states[((long)step * BATCH + b) * NHID + c] = to_bf16(hv);
}

// ---------------------------------------------------------------------------
extern "C" void kernel_launch(void* const* d_in, const int* in_sizes, int n_in,
                              void* d_out, int out_size, void* d_ws, size_t ws_size,
                              hipStream_t stream) {
    const int*   obs    = (const int*)  d_in[0];
    const float* embW   = (const float*)d_in[1];
    const float* q_w    = (const float*)d_in[2];
    const float* q_b    = (const float*)d_in[3];
    const float* q_g    = (const float*)d_in[4];
    const float* q_be   = (const float*)d_in[5];
    const float* int_w  = (const float*)d_in[6];
    const float* int_b  = (const float*)d_in[7];
    const float* int_g  = (const float*)d_in[8];
    const float* int_be = (const float*)d_in[9];
    const float* fin_w  = (const float*)d_in[10];
    const float* fin_b  = (const float*)d_in[11];
    const float* fin_g  = (const float*)d_in[12];
    const float* fin_be = (const float*)d_in[13];
    const float* dec_w  = (const float*)d_in[14];
    const float* dec_b  = (const float*)d_in[15];
    const float* h0     = (const float*)d_in[16];
    const float* k0     = (const float*)d_in[17];
    const float* v0     = (const float*)d_in[18];
    float* logits = (float*)d_out;

    char* p = (char*)d_ws;
    auto carve = [&](size_t bytes) -> void* {
        void* r = (void*)p;
        p += (bytes + 255) & ~(size_t)255;
        return r;
    };

    __bf16* qw_bf    = (__bf16*)carve((size_t)NHID * 2048 * 2);
    __bf16* iw_bf    = (__bf16*)carve((size_t)4 * NHID * 4 * NHID * 2);
    __bf16* fw_bf    = (__bf16*)carve((size_t)3 * NHID * 4 * NHID * 2);
    __bf16* dw_bf    = (__bf16*)carve((size_t)NTOKEN * NHID * 2);
    __bf16* emb_bf   = (__bf16*)carve((size_t)SEQ * BATCH * NINP * 2);
    __bf16* states_b = (__bf16*)carve((size_t)SEQ * BATCH * NHID * 2);
    __bf16* xq_bf    = (__bf16*)carve((size_t)BATCH * 2048 * 2);
    __bf16* xi_bf    = (__bf16*)carve((size_t)BATCH * 4096 * 2);
    __bf16* inter_bf = (__bf16*)carve((size_t)BATCH * 4096 * 2);
    float*  keys     = (float*)carve((size_t)(SEQ + 1) * BATCH * NHID * 4);
    float*  values   = (float*)carve((size_t)(SEQ + 1) * BATCH * NHID * 4);
    float*  hidden   = (float*)carve((size_t)BATCH * NHID * 4);
    float*  qf       = (float*)carve((size_t)BATCH * NHID * 4);
    float*  pre      = (float*)carve((size_t)BATCH * 4096 * 4);
    float*  finf     = (float*)carve((size_t)BATCH * 3 * NHID * 4);

    // one-time: bf16 weights (afterwards resident in the 192MB L2), gather, init
    k_cvt_bf16<<<2048, 256, 0, stream>>>(q_w,  qw_bf, (long)NHID * 2048);
    k_cvt_bf16<<<4096, 256, 0, stream>>>(int_w, iw_bf, (long)4 * NHID * 4 * NHID);
    k_cvt_bf16<<<4096, 256, 0, stream>>>(fin_w, fw_bf, (long)3 * NHID * 4 * NHID);
    k_cvt_bf16<<<8192, 256, 0, stream>>>(dec_w, dw_bf, (long)NTOKEN * NHID);
    k_gather_emb<<<SEQ * BATCH, 256, 0, stream>>>(embW, obs, emb_bf);
    k_init<<<(BATCH * NHID) / 256, 256, 0, stream>>>(k0, v0, h0, keys, values, hidden);

    for (int i = 0; i < SEQ; ++i) {
        k_build_inputs<<<(BATCH * NHID) / 256, 256, 0, stream>>>(
            emb_bf + (long)i * BATCH * NINP, hidden, xq_bf, xi_bf);
        // q path: [32,2048] x [2048,1024]
        k_gemm_wmma<<<dim3(NHID / 256, 1), 256, 0, stream>>>(xq_bf, qw_bf, q_b, pre, 2048, NHID);
        k_ln_relu<<<BATCH, 256, 0, stream>>>(pre, q_g, q_be, qf, xi_bf, NHID, 4096, 1024);
        // attention over slots 0..i
        k_attn<<<BATCH * NHEADS, 256, 0, stream>>>(qf, keys, values, xi_bf, i);
        // intermediate: [32,4096] x [4096,4096]
        k_gemm_wmma<<<dim3(4 * NHID / 256, 1), 256, 0, stream>>>(xi_bf, iw_bf, int_b, pre, 4096, 4 * NHID);
        k_ln_relu<<<BATCH, 256, 0, stream>>>(pre, int_g, int_be, nullptr, inter_bf, 4 * NHID, 4096, 0);
        // final: [32,4096] x [4096,3072]
        k_gemm_wmma<<<dim3(3 * NHID / 256, 1), 256, 0, stream>>>(inter_bf, fw_bf, fin_b, pre, 4096, 3 * NHID);
        k_ln_relu<<<BATCH, 256, 0, stream>>>(pre, fin_g, fin_be, finf, nullptr, 3 * NHID, 0, 0);
        k_commit<<<(BATCH * NHID) / 256, 256, 0, stream>>>(finf, keys, values, hidden, states_b, i);
    }

    // decoder: [8192,1024] x [1024,32000]  (~537 GFLOP, WMMA-tiled)
    k_gemm_wmma<<<dim3(NTOKEN / 256, (SEQ * BATCH) / 32), 256, 0, stream>>>(
        states_b, dw_bf, dec_b, logits, NHID, NTOKEN);
}
